// SparseResNet_68659347194198
// MI455X (gfx1250) — compile-verified, matrix operand order checked by reading, not run on hardware
//
#include <hip/hip_runtime.h>
#include <hip/hip_bf16.h>

typedef __attribute__((ext_vector_type(16))) __bf16        v16bf;
typedef __attribute__((ext_vector_type(8)))  float         v8f;
typedef __attribute__((ext_vector_type(8)))  unsigned int  v8u;

// float -> bf16 raw bits (round-to-nearest-even)
static __device__ inline unsigned short f2bf(float f) {
    unsigned u = __builtin_bit_cast(unsigned, f);
    u += 0x7FFFu + ((u >> 16) & 1u);
    return (unsigned short)(u >> 16);
}
static __device__ inline float bf2f(unsigned short h) {
    return __builtin_bit_cast(float, (unsigned)h << 16);
}

// ---------------------------------------------------------------------------
// Implicit-GEMM conv via v_wmma_f32_16x16x32_bf16, bf16 activations/weights,
// double-buffered LDS software pipeline (1 barrier per K-chunk, global loads
// of chunk k+1 overlap the WMMAs of chunk k).
//   X  : [B, Hin, Win, Cin]   bf16 activations
//   Wt : [KS*KS, Cout, Cin]   bf16 weights (k-contiguous, pre-transposed)
//   Out: [M, Cout] bf16,  M = B*Hout*Wout
//   epilogue: o = (acc*bnw[n] + bnb[n]) * mask[m]  (+ res[m,n])  (relu?)
// 256 threads = 8 waves; 64x64 block tile; each wave: 16 rows x 32 cols.
// Cin is a power-of-two multiple of 32 (64..1024) -> kcShift/kcMask.
// ---------------------------------------------------------------------------
__global__ __launch_bounds__(256) void wmma_conv_gemm(
    const unsigned short* __restrict__ X, const unsigned short* __restrict__ Wt,
    unsigned short* __restrict__ Out,
    const float* __restrict__ bnw, const float* __restrict__ bnb,
    const float* __restrict__ maskp, const unsigned short* __restrict__ resp,
    int M, int N, int Cin, int Hin, int Win, int Hout, int Wout,
    int stride, int KS, int pad, int relu, int kcShift, int nit)
{
    __shared__ unsigned short As[2][64 * 32];   // [row][k]
    __shared__ unsigned short Bs[2][64 * 32];   // [n][k]

    const int tid  = threadIdx.x;
    const int lane = tid & 31;
    const int wv   = tid >> 5;          // 0..7
    const int wm   = wv >> 1;           // 0..3 : row block of 16
    const int wn   = wv & 1;            // 0..1 : col block of 32
    const int n0   = blockIdx.x * 64;
    const int m0   = blockIdx.y * 64;
    const int HW   = Hout * Wout;
    const int kcMask = (1 << kcShift) - 1;

    // cooperative fill: each thread fills 8 contiguous k's of one row/col
    const int fr = tid >> 2;            // 0..63
    const int fk = (tid & 3) * 8;       // 0,8,16,24

    const int am = m0 + fr;
    const bool arow_ok = am < M;
    int ab = 0, aho = 0, awo = 0;
    if (arow_ok) { ab = am / HW; int r = am - ab * HW; aho = r / Wout; awo = r - aho * Wout; }
    const int gn = n0 + fr;             // N is a multiple of 64

    // stage chunk `it` into registers
    auto stage = [&](int it, uint4& va, uint4& vb) {
        const int tap = it >> kcShift;
        const int kc  = (it & kcMask) << 5;
        const int ky  = tap / KS;
        const int kx  = tap - ky * KS;
        const int hi  = aho * stride + ky - pad;
        const int wi  = awo * stride + kx - pad;
        const bool avalid = arow_ok && ((unsigned)hi < (unsigned)Hin) &&
                                       ((unsigned)wi < (unsigned)Win);
        va = make_uint4(0u, 0u, 0u, 0u);
        if (avalid)
            va = *(const uint4*)(X + (((size_t)ab * Hin + hi) * Win + wi) * (size_t)Cin
                                   + kc + fk);
        vb = *(const uint4*)(Wt + ((size_t)tap * N + gn) * (size_t)Cin + kc + fk);
    };

    v8f acc0 = {}, acc1 = {};
    const int khalf = lane >> 4;
    const int rowA  = wm * 16 + (lane & 15);
    const int colB0 = wn * 32 + (lane & 15);
    const int colB1 = colB0 + 16;

    // prologue: stage chunk 0 into buffer 0
    {
        uint4 va, vb;
        stage(0, va, vb);
        *(uint4*)(&As[0][fr * 32 + fk]) = va;
        *(uint4*)(&Bs[0][fr * 32 + fk]) = vb;
    }
    __syncthreads();

    for (int it = 0; it < nit; ++it) {
        const int cb = it & 1;
        const bool have_next = (it + 1) < nit;
        uint4 na, nb;
        if (have_next) stage(it + 1, na, nb);     // global loads in flight

        // ---- fragments from current buffer (paired-K layout) ----
        const unsigned short* ap = &As[cb][rowA * 32 + khalf * 8];
        uint4 al = *(const uint4*)(ap);
        uint4 ah = *(const uint4*)(ap + 16);
        const unsigned short* b0p = &Bs[cb][colB0 * 32 + khalf * 8];
        uint4 b0l = *(const uint4*)(b0p);
        uint4 b0h = *(const uint4*)(b0p + 16);
        const unsigned short* b1p = &Bs[cb][colB1 * 32 + khalf * 8];
        uint4 b1l = *(const uint4*)(b1p);
        uint4 b1h = *(const uint4*)(b1p + 16);

        v8u au, bu0, bu1;
        au[0] = al.x;  au[1] = al.y;  au[2] = al.z;  au[3] = al.w;
        au[4] = ah.x;  au[5] = ah.y;  au[6] = ah.z;  au[7] = ah.w;
        bu0[0] = b0l.x; bu0[1] = b0l.y; bu0[2] = b0l.z; bu0[3] = b0l.w;
        bu0[4] = b0h.x; bu0[5] = b0h.y; bu0[6] = b0h.z; bu0[7] = b0h.w;
        bu1[0] = b1l.x; bu1[1] = b1l.y; bu1[2] = b1l.z; bu1[3] = b1l.w;
        bu1[4] = b1h.x; bu1[5] = b1h.y; bu1[6] = b1h.z; bu1[7] = b1h.w;

        v16bf af  = __builtin_bit_cast(v16bf, au);
        v16bf bf0 = __builtin_bit_cast(v16bf, bu0);
        v16bf bf1 = __builtin_bit_cast(v16bf, bu1);
        acc0 = __builtin_amdgcn_wmma_f32_16x16x32_bf16(
                   false, af, false, bf0, (short)0, acc0, false, false);
        acc1 = __builtin_amdgcn_wmma_f32_16x16x32_bf16(
                   false, af, false, bf1, (short)0, acc1, false, false);

        // ---- store next chunk into the other buffer, then one barrier ----
        if (have_next) {
            const int nbuf = cb ^ 1;
            *(uint4*)(&As[nbuf][fr * 32 + fk]) = na;
            *(uint4*)(&Bs[nbuf][fr * 32 + fk]) = nb;
        }
        __syncthreads();
    }

    // ---- epilogue ----
    const int nA = n0 + wn * 32 + (lane & 15);
    const int nB = nA + 16;
    const int mbase = m0 + wm * 16 + khalf * 8;
    const float bwA = bnw[nA], bbA = bnb[nA];
    const float bwB = bnw[nB], bbB = bnb[nB];
#pragma unroll
    for (int r = 0; r < 8; ++r) {
        const int m = mbase + r;
        if (m < M) {
            const float mk = maskp[m];
            float oA = (acc0[r] * bwA + bbA) * mk;
            float oB = (acc1[r] * bwB + bbB) * mk;
            if (resp) {
                oA += bf2f(resp[(size_t)m * N + nA]);
                oB += bf2f(resp[(size_t)m * N + nB]);
            }
            if (relu) { oA = fmaxf(oA, 0.0f); oB = fmaxf(oB, 0.0f); }
            Out[(size_t)m * N + nA] = f2bf(oA);
            Out[(size_t)m * N + nB] = f2bf(oB);
        }
    }
}

// ---------------------------------------------------------------------------
// Weight transform: HWIO f32 [taps,Cin,Cout] -> [taps,Cout,Cin] bf16
// ---------------------------------------------------------------------------
__global__ void wconv(const float* __restrict__ w, unsigned short* __restrict__ wt,
                      int taps, int Cin, int Cout)
{
    int i = blockIdx.x * blockDim.x + threadIdx.x;
    if (i >= taps * Cin * Cout) return;
    const int ci = i % Cin;
    int t = i / Cin;
    const int co = t % Cout;
    const int tap = t / Cout;
    wt[i] = f2bf(w[((size_t)tap * Cin + ci) * Cout + co]);
}

// ---------------------------------------------------------------------------
// Stem: 7x7 s2 VALID conv over (x*mask), Cin=3, Cout=64, relu(affine*m1), bf16 out
// ---------------------------------------------------------------------------
__global__ void stem_conv(const float* __restrict__ x,
                          const unsigned char* __restrict__ mask0,
                          const float* __restrict__ w,
                          const float* __restrict__ bnw, const float* __restrict__ bnb,
                          const float* __restrict__ m1, unsigned short* __restrict__ out,
                          int total)
{
    int i = blockIdx.x * blockDim.x + threadIdx.x;
    if (i >= total) return;
    const int c = i & 63;
    int t = i >> 6;
    const int wo = t % 251; t /= 251;
    const int ho = t % 251;
    const int b  = t / 251;
    float acc = 0.0f;
    for (int ky = 0; ky < 7; ++ky)
        for (int kx = 0; kx < 7; ++kx) {
            const int hi = ho * 2 + ky, wi = wo * 2 + kx;
            const size_t pix = ((size_t)b * 507 + hi) * 507 + wi;
            if (mask0[pix]) {
                const float* xp = x + pix * 3;
                const float* wp = w + ((ky * 7 + kx) * 3) * 64 + c;
                acc += xp[0] * wp[0] + xp[1] * wp[64] + xp[2] * wp[128];
            }
        }
    float o = (acc * bnw[c] + bnb[c]) * m1[((size_t)b * 251 + ho) * 251 + wo];
    out[i] = f2bf(fmaxf(o, 0.0f));
}

__global__ void mask_to_float(const unsigned char* __restrict__ in,
                              float* __restrict__ out, int total)
{
    int i = blockIdx.x * blockDim.x + threadIdx.x;
    if (i < total) out[i] = in[i] ? 1.0f : 0.0f;
}

// reduce_window max, init 0.0 (matches reference)
__global__ void pool_mask(const float* __restrict__ in, float* __restrict__ out,
                          int B, int Hin, int Win, int Hout, int Wout, int k, int s)
{
    int i = blockIdx.x * blockDim.x + threadIdx.x;
    if (i >= B * Hout * Wout) return;
    const int wo = i % Wout;
    int t = i / Wout;
    const int ho = t % Hout;
    const int b  = t / Hout;
    float mx = 0.0f;
    for (int dy = 0; dy < k; ++dy)
        for (int dx = 0; dx < k; ++dx)
            mx = fmaxf(mx, in[((size_t)b * Hin + ho * s + dy) * Win + wo * s + dx]);
    out[i] = mx;
}

// 3x3 s2 VALID channel-wise maxpool on bf16, init 0.0
__global__ void maxpool_c(const unsigned short* __restrict__ in,
                          unsigned short* __restrict__ out,
                          int B, int Hin, int Win, int C, int Hout, int Wout)
{
    int i = blockIdx.x * blockDim.x + threadIdx.x;
    if (i >= B * Hout * Wout * C) return;
    const int c = i % C;
    int t = i / C;
    const int wo = t % Wout; t /= Wout;
    const int ho = t % Hout;
    const int b  = t / Hout;
    float mx = 0.0f;
    for (int dy = 0; dy < 3; ++dy)
        for (int dx = 0; dx < 3; ++dx)
            mx = fmaxf(mx, bf2f(in[(((size_t)b * Hin + ho * 2 + dy) * Win + wo * 2 + dx) * C + c]));
    out[i] = f2bf(mx);
}

__global__ void subsample2(const float* __restrict__ in, float* __restrict__ out,
                           int B, int Hin, int Win, int Hout, int Wout)
{
    int i = blockIdx.x * blockDim.x + threadIdx.x;
    if (i >= B * Hout * Wout) return;
    const int wo = i % Wout;
    int t = i / Wout;
    const int ho = t % Hout;
    const int b  = t / Hout;
    out[i] = in[((size_t)b * Hin + ho * 2) * Win + wo * 2];
}

// [4,32,32,1024] NHWC bf16 -> [4,1024,32,31] NCHW f32, crop last column
__global__ void out_tr(const unsigned short* __restrict__ in, float* __restrict__ out,
                       int total)
{
    int i = blockIdx.x * blockDim.x + threadIdx.x;
    if (i >= total) return;
    const int w = i % 31;
    int t = i / 31;
    const int h = t % 32; t /= 32;
    const int c = t % 1024;
    const int b = t / 1024;
    out[i] = bf2f(in[(((size_t)b * 32 + h) * 32 + w) * 1024 + c]);
}

// ---------------------------------------------------------------------------
extern "C" void kernel_launch(void* const* d_in, const int* in_sizes, int n_in,
                              void* d_out, int out_size, void* d_ws, size_t ws_size,
                              hipStream_t stream)
{
    (void)in_sizes; (void)n_in; (void)ws_size;
    const float* x = (const float*)d_in[0];
    const unsigned char* mask = (const unsigned char*)d_in[1];

    // ---- unpack params (jax pytree order: dict keys sorted) ----
    int idx = 2;
    const float* r1_bnb  = (const float*)d_in[idx++];
    const float* r1_bnw  = (const float*)d_in[idx++];
    const float* r1_conv = (const float*)d_in[idx++];

    struct Blk { const float *bn1b,*bn1w,*bn2b,*bn2w,*bn3b,*bn3w,*c1,*c2,*c3,*dbnb,*dbnw,*dc; };
    Blk blocks[13];
    const int counts[3] = {3, 4, 6};
    int nb = 0;
    for (int s = 0; s < 3; ++s)
        for (int i = 0; i < counts[s]; ++i) {
            Blk b{};
            b.bn1b = (const float*)d_in[idx++]; b.bn1w = (const float*)d_in[idx++];
            b.bn2b = (const float*)d_in[idx++]; b.bn2w = (const float*)d_in[idx++];
            b.bn3b = (const float*)d_in[idx++]; b.bn3w = (const float*)d_in[idx++];
            b.c1   = (const float*)d_in[idx++];
            b.c2   = (const float*)d_in[idx++];
            b.c3   = (const float*)d_in[idx++];
            if (i == 0) {
                b.dbnb = (const float*)d_in[idx++];
                b.dbnw = (const float*)d_in[idx++];
                b.dc   = (const float*)d_in[idx++];
            }
            blocks[nb++] = b;
        }

    // ---- workspace carve (bytes, 256B aligned) ----
    char* base = (char*)d_ws;
    size_t off = 0;
    auto alloc = [&](size_t bytes) { void* p = base + off; off += (bytes + 255) & ~(size_t)255; return p; };
    unsigned short* bufA = (unsigned short*)alloc((size_t)16200000 * 2);
    unsigned short* bufB = (unsigned short*)alloc((size_t)16200000 * 2);
    unsigned short* R    = (unsigned short*)alloc((size_t)16200000 * 2);
    unsigned short* T1   = (unsigned short*)alloc((size_t)4100000 * 2);
    unsigned short* T2   = (unsigned short*)alloc((size_t)4100000 * 2);
    unsigned short* wtar = (unsigned short*)alloc((size_t)9000000 * 2);
    float* m0f = (float*)alloc((size_t)4 * 507 * 507 * 4);
    float* m1  = (float*)alloc((size_t)4 * 251 * 251 * 4);
    float* m2  = (float*)alloc((size_t)4 * 125 * 125 * 4);
    float* m3  = (float*)alloc((size_t)4 * 63 * 63 * 4);
    float* m4  = (float*)alloc((size_t)4 * 32 * 32 * 4);

    size_t wt_off = 0;
    auto conv_w = [&](const float* w, int taps, int Cin, int Cout) -> const unsigned short* {
        unsigned short* dst = wtar + wt_off;
        wt_off += (size_t)taps * Cin * Cout;
        const int total = taps * Cin * Cout;
        wconv<<<(total + 255) / 256, 256, 0, stream>>>(w, dst, taps, Cin, Cout);
        return dst;
    };

    auto gemm = [&](const unsigned short* X, const unsigned short* Wt, unsigned short* O,
                    const float* bw, const float* bb, const float* mo,
                    const unsigned short* res,
                    int Hin, int Hout, int Cin, int Cout, int stride, int ks, int relu) {
        const int M = 4 * Hout * Hout;
        int kcShift = 0;
        while ((32 << kcShift) < Cin) ++kcShift;       // Cin = 32 << kcShift
        const int nit = ks * ks * (Cin >> 5);
        dim3 g(Cout / 64, (M + 63) / 64);
        wmma_conv_gemm<<<g, 256, 0, stream>>>(X, Wt, O, bw, bb, mo, res,
                                              M, Cout, Cin, Hin, Hin, Hout, Hout,
                                              stride, ks, (ks == 3) ? 1 : 0, relu,
                                              kcShift, nit);
    };

    // ---- stem ----
    int t0 = 4 * 507 * 507;
    mask_to_float<<<(t0 + 255) / 256, 256, 0, stream>>>(mask, m0f, t0);
    int t1 = 4 * 251 * 251;
    pool_mask<<<(t1 + 255) / 256, 256, 0, stream>>>(m0f, m1, 4, 507, 507, 251, 251, 7, 2);
    int ts = 4 * 251 * 251 * 64;
    stem_conv<<<(ts + 255) / 256, 256, 0, stream>>>(x, mask, r1_conv, r1_bnw, r1_bnb,
                                                    m1, bufA, ts);
    int t2 = 4 * 125 * 125;
    pool_mask<<<(t2 + 255) / 256, 256, 0, stream>>>(m1, m2, 4, 251, 251, 125, 125, 3, 2);
    int tp = 4 * 125 * 125 * 64;
    maxpool_c<<<(tp + 255) / 256, 256, 0, stream>>>(bufA, bufB, 4, 251, 251, 64, 125, 125);
    int t3 = 4 * 63 * 63;
    subsample2<<<(t3 + 255) / 256, 256, 0, stream>>>(m2, m3, 4, 125, 125, 63, 63);
    int t4 = 4 * 32 * 32;
    subsample2<<<(t4 + 255) / 256, 256, 0, stream>>>(m3, m4, 4, 63, 63, 32, 32);

    // ---- bottleneck stages ----
    struct Stage { int n, cin, cout, inner, s0, Hin, Hout; const float* mo; };
    Stage stages[3] = {
        {3,  64,  256,  64, 1, 125, 125, m2},
        {4, 256,  512, 128, 2, 125,  63, m3},
        {6, 512, 1024, 256, 2,  63,  32, m4},
    };

    unsigned short* cur = bufB;
    unsigned short* oth = bufA;
    int bi = 0;
    for (int s = 0; s < 3; ++s) {
        Stage& st = stages[s];
        for (int i = 0; i < st.n; ++i) {
            Blk& b = blocks[bi++];
            const int stride = (i == 0) ? st.s0 : 1;
            const int Hin    = (i == 0) ? st.Hin : st.Hout;
            const int cin    = (i == 0) ? st.cin : st.cout;
            const unsigned short* res;
            if (i == 0) {
                const unsigned short* wd = conv_w(b.dc, 1, cin, st.cout);
                gemm(cur, wd, R, b.dbnw, b.dbnb, st.mo, nullptr,
                     Hin, st.Hout, cin, st.cout, st.s0, 1, 0);
                res = R;
            } else {
                res = cur;   // identity residual
            }
            const unsigned short* w1 = conv_w(b.c1, 1, cin, st.inner);
            gemm(cur, w1, T1, b.bn1w, b.bn1b, st.mo, nullptr,
                 Hin, st.Hout, cin, st.inner, stride, 1, 1);
            const unsigned short* w2 = conv_w(b.c2, 9, st.inner, st.inner);
            gemm(T1, w2, T2, b.bn2w, b.bn2b, st.mo, nullptr,
                 st.Hout, st.Hout, st.inner, st.inner, 1, 3, 1);
            const unsigned short* w3 = conv_w(b.c3, 1, st.inner, st.cout);
            gemm(T2, w3, oth, b.bn3w, b.bn3b, st.mo, res,
                 st.Hout, st.Hout, st.inner, st.cout, 1, 1, 1);
            unsigned short* t = cur; cur = oth; oth = t;
        }
    }

    // ---- final transpose + crop ----
    out_tr<<<(out_size + 255) / 256, 256, 0, stream>>>(cur, (float*)d_out, out_size);
}
